// MomentExtraction_52321291600116
// MI455X (gfx1250) — matrix operand order repaired and checked by reading, not compiled, not run on hardware
//
#include <hip/hip_runtime.h>
#include <hip/hip_bf16.h>

typedef __attribute__((ext_vector_type(16))) __bf16 v16bf;
typedef __attribute__((ext_vector_type(8)))  float  v8f;

#define NUM_CLS   19
#define KPAD      32
#define CCH       256
#define NPIX      65536          // 256*256 pixels per image
#define BATCH     4
#define NCG       16             // channel groups of 16
#define SPLIT     16             // pixel-dim split per (b, cg)
#define EPSF      1e-5f

// ---- ws layout (floats): [0..32767]=S, [32768..65535]=SS, then 128 ints of counts
#define WS_S_ELEMS   (BATCH * KPAD * CCH)        // 32768
#define WS_TOT_WORDS (2 * WS_S_ELEMS + BATCH * KPAD)  // 65664 32-bit words

// -------------------------------------------------------------------- zero ws
__global__ __launch_bounds__(256) void zero_ws_kernel(unsigned* ws, int n) {
    int i = blockIdx.x * 256 + threadIdx.x;
    if (i < n) ws[i] = 0u;
}

// ------------------------------------------------------------- label histogram
__global__ __launch_bounds__(256) void hist_kernel(const int* __restrict__ y,
                                                   int* __restrict__ cnt) {
    __shared__ int lh[KPAD];
    int tid = threadIdx.x;
    if (tid < KPAD) lh[tid] = 0;
    __syncthreads();
    int b   = blockIdx.x >> 6;            // 64 blocks per batch
    int off = (blockIdx.x & 63) * 1024;   // 1024 labels per block
    const int* yp = y + b * NPIX + off;
#pragma unroll
    for (int j = 0; j < 4; ++j) {
        int lbl = yp[tid + j * 256] & 31;
        atomicAdd(&lh[lbl], 1);
    }
    __syncthreads();
    if (tid < KPAD) atomicAdd(&cnt[b * KPAD + tid], lh[tid]);
}

// ----------------------------------------------------- main WMMA moment kernel
// grid = BATCH*NCG*SPLIT = 1024 blocks of 256 threads (8 waves of 32)
__global__ __launch_bounds__(256) void moments_kernel(const float* __restrict__ x,
                                                      const int*   __restrict__ y,
                                                      float* __restrict__ wsS,
                                                      float* __restrict__ wsSS) {
    const int wg    = blockIdx.x;
    const int split = wg & (SPLIT - 1);
    const int cg    = (wg >> 4) & (NCG - 1);
    const int b     = wg >> 8;

    const int tid  = threadIdx.x;
    const int w    = tid >> 5;       // wave id 0..7
    const int lane = tid & 31;
    const int half = lane >> 4;      // 0: K 0..7 & 16..23 (A) / pix 0..7 & 16..23 (B)
    const int m    = lane & 15;      // class-in-tile / channel-in-group

    const int* yb  = y + b * NPIX;
    const float* xch = x + ((size_t)(b * CCH + cg * 16 + m)) * NPIX;

    v8f s0 = {0,0,0,0,0,0,0,0};      // S,  classes 0..15
    v8f s1 = {0,0,0,0,0,0,0,0};      // S,  classes 16..31
    v8f q0 = {0,0,0,0,0,0,0,0};      // SS, classes 0..15
    v8f q1 = {0,0,0,0,0,0,0,0};      // SS, classes 16..31

    const int sliceBase = split * (NPIX / SPLIT) + w * (NPIX / SPLIT / 8); // 4096/512

    for (int chnk = 0; chnk < 16; ++chnk) {
        const int p0 = sliceBase + chnk * 32;

        // ---- labels -> onehot A fragments (A layout: half0 K={0..7,16..23})
        const int4* yp1 = (const int4*)(yb + p0 + 8 * half);
        const int4* yp2 = (const int4*)(yb + p0 + 16 + 8 * half);
        int4 la = yp1[0], lb = yp1[1], lc = yp2[0], ld = yp2[1];
        int lbls[16] = { la.x, la.y, la.z, la.w,  lb.x, lb.y, lb.z, lb.w,
                         lc.x, lc.y, lc.z, lc.w,  ld.x, ld.y, ld.z, ld.w };
        v16bf a0, a1;
#pragma unroll
        for (int i = 0; i < 16; ++i) {
            a0[i] = (lbls[i] == m)        ? (__bf16)1.0f : (__bf16)0.0f;
            a1[i] = (lbls[i] == (m + 16)) ? (__bf16)1.0f : (__bf16)0.0f;
        }

        // ---- x -> B fragments (lane holds pixels {8h..8h+7, 16+8h..16+8h+7} of channel m)
        const float4* xp  = (const float4*)(xch + p0 + 8 * half);
        const float4* xp2 = (const float4*)(xch + p0 + 16 + 8 * half);
        float4 f0 = xp[0], f1 = xp[1], f2 = xp2[0], f3 = xp2[1];
        if (chnk + 1 < 16) __builtin_prefetch(xch + p0 + 32 + 16 * half, 0, 1);
        float xv[16] = { f0.x, f0.y, f0.z, f0.w,  f1.x, f1.y, f1.z, f1.w,
                         f2.x, f2.y, f2.z, f2.w,  f3.x, f3.y, f3.z, f3.w };
        v16bf bh, bl, ch2, cl2;   // hi/lo split of x and of x*x
#pragma unroll
        for (int i = 0; i < 16; ++i) {
            float v  = xv[i];
            __bf16 h = (__bf16)v;
            bh[i]  = h;
            bl[i]  = (__bf16)(v - (float)h);
            float v2  = v * v;
            __bf16 h2 = (__bf16)v2;
            ch2[i] = h2;
            cl2[i] = (__bf16)(v2 - (float)h2);
        }

        // ---- 8 bf16 WMMAs: (2 class tiles) x (S,SS) x (hi,lo)
        s0 = __builtin_amdgcn_wmma_f32_16x16x32_bf16(false, a0, false, bh,  (short)0, s0, false, false);
        s0 = __builtin_amdgcn_wmma_f32_16x16x32_bf16(false, a0, false, bl,  (short)0, s0, false, false);
        s1 = __builtin_amdgcn_wmma_f32_16x16x32_bf16(false, a1, false, bh,  (short)0, s1, false, false);
        s1 = __builtin_amdgcn_wmma_f32_16x16x32_bf16(false, a1, false, bl,  (short)0, s1, false, false);
        q0 = __builtin_amdgcn_wmma_f32_16x16x32_bf16(false, a0, false, ch2, (short)0, q0, false, false);
        q0 = __builtin_amdgcn_wmma_f32_16x16x32_bf16(false, a0, false, cl2, (short)0, q0, false, false);
        q1 = __builtin_amdgcn_wmma_f32_16x16x32_bf16(false, a1, false, ch2, (short)0, q1, false, false);
        q1 = __builtin_amdgcn_wmma_f32_16x16x32_bf16(false, a1, false, cl2, (short)0, q1, false, false);
    }

    // ---- cross-wave reduction in LDS, then one set of global fp32 atomics
    __shared__ float red[8][1024];
#pragma unroll
    for (int r = 0; r < 8; ++r) {
        red[w][0 * 256 + r * 32 + lane] = s0[r];
        red[w][1 * 256 + r * 32 + lane] = s1[r];
        red[w][2 * 256 + r * 32 + lane] = q0[r];
        red[w][3 * 256 + r * 32 + lane] = q1[r];
    }
    __syncthreads();
#pragma unroll
    for (int q = 0; q < 4; ++q) {
        int idx = q * 256 + tid;
        float acc = 0.f;
#pragma unroll
        for (int ww = 0; ww < 8; ++ww) acc += red[ww][idx];
        // decode C/D fragment layout: VGPR r, lanes0-15 -> M=r, lanes16-31 -> M=r+8
        int f   = idx >> 8;          // 0:S-t0, 1:S-t1, 2:SS-t0, 3:SS-t1
        int r   = (idx >> 5) & 7;
        int l   = idx & 31;
        int cls = r + 8 * (l >> 4) + 16 * (f & 1);
        int c   = cg * 16 + (l & 15);
        float* dst = (f >> 1) ? wsSS : wsS;
        atomicAdd(dst + (size_t)(b * KPAD + cls) * CCH + c, acc);
    }
}

// ------------------------------------------------------------------- finalize
__global__ __launch_bounds__(256) void finalize_kernel(const float* __restrict__ wsS,
                                                       const float* __restrict__ wsSS,
                                                       const int*   __restrict__ cnts,
                                                       float* __restrict__ out) {
    int t = blockIdx.x * 256 + threadIdx.x;
    if (t >= BATCH * NUM_CLS * CCH) return;
    int c = t & 255;
    int k = (t >> 8) % NUM_CLS;
    int b = t / (NUM_CLS * CCH);

    float cnt = (float)cnts[b * KPAD + k];
    float s   = wsS [(size_t)(b * KPAD + k) * CCH + c];
    float ss  = wsSS[(size_t)(b * KPAD + k) * CCH + c];

    bool  valid = cnt > 6.0f;
    float mean  = s / fmaxf(cnt, 1.0f);
    float denom = fmaxf(cnt - 1.0f, 1.0f);
    float var   = fmaxf((ss - cnt * mean * mean) / denom, 0.0f);
    float stdv  = sqrtf(valid ? var : 1.0f) + EPSF;

    int oidx = (b * NUM_CLS + k) * CCH + c;
    out[oidx]                           = valid ? mean : 0.0f;  // means [B,K,C]
    out[BATCH * NUM_CLS * CCH + oidx]   = valid ? stdv : 0.0f;  // stds  [B,K,C]
    if (c == 0)
        out[2 * BATCH * NUM_CLS * CCH + b * NUM_CLS + k] = valid ? 1.0f : 0.0f; // valid
}

// -------------------------------------------------------------------- launch
extern "C" void kernel_launch(void* const* d_in, const int* in_sizes, int n_in,
                              void* d_out, int out_size, void* d_ws, size_t ws_size,
                              hipStream_t stream) {
    const float* x = (const float*)d_in[0];
    const int*   y = (const int*)d_in[1];
    float* out = (float*)d_out;

    float* wsS  = (float*)d_ws;
    float* wsSS = wsS + WS_S_ELEMS;
    int*   wsC  = (int*)(wsS + 2 * WS_S_ELEMS);

    zero_ws_kernel<<<(WS_TOT_WORDS + 255) / 256, 256, 0, stream>>>((unsigned*)d_ws, WS_TOT_WORDS);
    hist_kernel<<<BATCH * 64, 256, 0, stream>>>(y, wsC);
    moments_kernel<<<BATCH * NCG * SPLIT, 256, 0, stream>>>(x, y, wsS, wsSS);
    finalize_kernel<<<(BATCH * NUM_CLS * CCH + 255) / 256, 256, 0, stream>>>(wsS, wsSS, wsC, out);
}
